// LDGCNNClassification_22479858828028
// MI455X (gfx1250) — matrix operand order repaired and checked by reading, not compiled, not run on hardware
//
#include <hip/hip_runtime.h>
#include <hip/hip_bf16.h>
#include <stdint.h>

typedef __attribute__((ext_vector_type(16))) __bf16       bf16x16;
typedef __attribute__((ext_vector_type(8)))  float        f32x8;
typedef __attribute__((ext_vector_type(2)))  float        f32x2;
typedef __attribute__((ext_vector_type(8)))  unsigned int u32x8;
typedef __attribute__((ext_vector_type(4)))  unsigned int u32x4;

#define BN_INV 0.99999500003749973f
#define SLOPE  0.2f
#define NPTS   1024
#define NB     8
#define NROWS  (NB * NPTS)
#define NEG_INF (-3.0e38f)

// ---------------------------------------------------------------------------
// squared norms per point
// ---------------------------------------------------------------------------
__global__ void sqn_kernel(const float* __restrict__ X, int ldx, int C,
                           float* __restrict__ sqn) {
    int t = blockIdx.x * blockDim.x + threadIdx.x;
    if (t >= NROWS) return;
    const float* row = X + (size_t)t * ldx;
    float s = 0.f;
    for (int c = 0; c < C; ++c) s += row[c] * row[c];
    sqn[t] = s;
}

// ---------------------------------------------------------------------------
// distance matrix D[b][n][m] = 2*x_n.x_m - |x_n|^2 - |x_m|^2 via f32 WMMA
// one 16x16 tile per wave; block = 4 waves; unguarded main K loop + tail
// ---------------------------------------------------------------------------
__global__ void dist_kernel(const float* __restrict__ X, int ldx, int C,
                            const float* __restrict__ sqn,
                            float* __restrict__ D) {
    int wid  = threadIdx.x >> 5;
    int lane = threadIdx.x & 31;
    int mt = blockIdx.x * 4 + wid;   // 0..63
    int nt = blockIdx.y;             // 0..63
    int b  = blockIdx.z;
    int n0 = nt * 16, m0 = mt * 16;
    int half = lane >> 4, l15 = lane & 15;
    const float* Xb = X + (size_t)b * NPTS * ldx;
    // A 16x4 f32: lane row = l15, VGPR i holds K = c0 + 2*half + i
    const float* pa = Xb + (size_t)(n0 + l15) * ldx + half * 2;
    const float* pb = Xb + (size_t)(m0 + l15) * ldx + half * 2;
    f32x8 acc = {};
    int Cmain = C & ~3;
    int c0 = 0;
    for (; c0 < Cmain; c0 += 4) {
        f32x2 a, bb;
        a[0]  = pa[c0]; a[1]  = pa[c0 + 1];
        bb[0] = pb[c0]; bb[1] = pb[c0 + 1];
        acc = __builtin_amdgcn_wmma_f32_16x16x4_f32(false, a, false, bb,
                                                    (short)0, acc, false, false);
    }
    if (C & 3) {                      // one guarded tail chunk
        f32x2 a = {}, bb = {};
#pragma unroll
        for (int i = 0; i < 2; ++i) {
            int k = Cmain + half * 2 + i;
            if (k < C) { a[i] = pa[Cmain + i]; bb[i] = pb[Cmain + i]; }
        }
        acc = __builtin_amdgcn_wmma_f32_16x16x4_f32(false, a, false, bb,
                                                    (short)0, acc, false, false);
    }
    const float* sq = sqn + b * NPTS;
    float sm = sq[m0 + l15];
    float* Db = D + (size_t)b * NPTS * NPTS;
#pragma unroll
    for (int r = 0; r < 8; ++r) {
        int n = n0 + r + half * 8;    // C/D: row = r + 8*half, col = l15
        float d = 2.f * acc[r] - sq[n] - sm;
        Db[(size_t)n * NPTS + m0 + l15] = d;
    }
}

// ---------------------------------------------------------------------------
// top-40 neighbor selection per row (sorted descending -> top-20 is a prefix)
// one wave per row, 8 rows per 256-thread block; row staged into LDS with
// GLOBAL_LOAD_ASYNC_TO_LDS_B128 (ASYNCcnt path, no VGPR round-trip)
// ---------------------------------------------------------------------------
__global__ void topk_kernel(const float* __restrict__ D, int* __restrict__ idx) {
    __shared__ __align__(16) float rowbuf[8][NPTS];
    int wid = threadIdx.x >> 5, lane = threadIdx.x & 31;
    int row = blockIdx.x * 8 + wid;
    float* rb = rowbuf[wid];
    const float* Dr = D + (size_t)row * NPTS;
    // async copy: 8 x b128 per lane (generic LDS ptr low 32 bits == LDS offset)
    unsigned ldsbase = (unsigned)(uintptr_t)&rb[0];
    for (int m4 = lane * 4; m4 < NPTS; m4 += 32 * 4) {
        unsigned laddr = ldsbase + (unsigned)m4 * 4u;
        unsigned long long gaddr = (unsigned long long)(uintptr_t)(Dr + m4);
        asm volatile("global_load_async_to_lds_b128 %0, %1, off"
                     :: "v"(laddr), "v"(gaddr) : "memory");
    }
    asm volatile("s_wait_asynccnt 0x0" ::: "memory");
    __syncthreads();
    int* out = idx + (size_t)row * 40;
    for (int it = 0; it < 40; ++it) {
        float bv = NEG_INF; int bi = 0;
        for (int m = lane; m < NPTS; m += 32) {
            float v = rb[m];
            if (v > bv || (v == bv && m < bi)) { bv = v; bi = m; }
        }
#pragma unroll
        for (int s = 16; s > 0; s >>= 1) {
            float ov = __shfl_xor(bv, s);
            int   oi = __shfl_xor(bi, s);
            if (ov > bv || (ov == bv && oi < bi)) { bv = ov; bi = oi; }
        }
        if (lane == 0) { out[it] = bi; rb[bi] = NEG_INF; }
        __syncthreads();
    }
}

// ---------------------------------------------------------------------------
// fragment loaders (bf16 WMMA, ISA VGPR layouts), vectorized b128 traffic
// ---------------------------------------------------------------------------
// B matrix KxN: lane = column n; elems [0..8) at K = half*8.., [8..16) at K = 16 + half*8..
// -> two 16B runs => 2 x global_load_b128
__device__ __forceinline__ bf16x16 load_bfrag(const __bf16* __restrict__ Bw,
                                              int N, int K, int ncol, int kk,
                                              int lane) {
    u32x8 u = {};
    if (ncol < N) {
        const __bf16* p = Bw + (size_t)ncol * K + kk + (lane >> 4) * 8;
        u32x4 lo = *reinterpret_cast<const u32x4*>(p);
        u32x4 hi = *reinterpret_cast<const u32x4*>(p + 16);
        u = (u32x8){lo[0], lo[1], lo[2], lo[3], hi[0], hi[1], hi[2], hi[3]};
    }
    return __builtin_bit_cast(bf16x16, u);
}

// A matrix 16x32 bf16 from LDS: lane = row m; elems [0..8) at p, [8..16) at p+16
// -> two 16B runs => 2 x ds_load_b128
__device__ __forceinline__ bf16x16 load_afrag_lds(const __bf16* fA, int ldA,
                                                  int jrow, int kk, int half) {
    const __bf16* p = fA + (size_t)jrow * ldA + kk + half * 8;
    u32x4 lo = *reinterpret_cast<const u32x4*>(p);
    u32x4 hi = *reinterpret_cast<const u32x4*>(p + 16);
    u32x8 u = {lo[0], lo[1], lo[2], lo[3], hi[0], hi[1], hi[2], hi[3]};
    return __builtin_bit_cast(bf16x16, u);
}

// A matrix from global f32: two contiguous 32B runs => 4 x global_load_b128,
// converted to bf16 in-register
__device__ __forceinline__ bf16x16 load_afrag_f32(const float* __restrict__ A,
                                                  int lda, int M, int mrow,
                                                  int kk, int half) {
    bf16x16 a = {};
    if (mrow < M) {
        const float* p = A + (size_t)mrow * lda + kk + half * 8;
        f32x8 lo = *reinterpret_cast<const f32x8*>(p);
        f32x8 hi = *reinterpret_cast<const f32x8*>(p + 16);
#pragma unroll
        for (int i = 0; i < 8; ++i) {
            a[i]     = (__bf16)lo[i];
            a[8 + i] = (__bf16)hi[i];
        }
    }
    return a;
}

// ---------------------------------------------------------------------------
// EdgeConv: one workgroup per (b,n); build [neigh-center ; center] in LDS as
// bf16, each wave computes a 16-output tile via bf16 WMMA over j-tiles, then
// LeakyReLU*BN_INV and masked max over k (in-register + shfl_xor(16)).
// CK is a power of two -> shift/mask indexing; neighbor ids cached in LDS.
// ---------------------------------------------------------------------------
__global__ void edgeconv_kernel(const float* __restrict__ Xin, int ldx, int C,
                                const int* __restrict__ knn, int K, int Jt,
                                const __bf16* __restrict__ Wb, int ckshift,
                                float* __restrict__ catOut, int catLD,
                                int oOff) {
    extern __shared__ char smem[];
    int CK = 1 << ckshift;
    int rn = blockIdx.x;             // 0..NROWS-1
    int b = rn >> 10;
    int C2 = 2 * C;
    int Jrows = Jt * 16;
    float* cen = (float*)smem;
    int*   jl  = (int*)(smem + (size_t)C * 4);
    __bf16* fA = (__bf16*)(smem + (((size_t)C * 4 + 192 + 63) & ~(size_t)63));
    const float* xc = Xin + (size_t)rn * ldx;
    int T = blockDim.x, tid = threadIdx.x;
    const int* kr = knn + (size_t)rn * 40;
    for (int c = tid; c < C; c += T) cen[c] = xc[c];
    for (int j = tid; j < Jrows; j += T) jl[j] = (j < K) ? kr[j] : 0;
    __syncthreads();
    int total = Jrows << ckshift;
    for (int e = tid; e < total; e += T) {
        int j = e >> ckshift, c = e & (CK - 1);
        float v = 0.f;
        if (j < K && c < C2) {
            if (c < C) {
                int nj = jl[j];
                v = Xin[(size_t)(b * NPTS + nj) * ldx + c] - cen[c];
            } else {
                v = cen[c - C];
            }
        }
        fA[((size_t)j << ckshift) + c] = (__bf16)v;
    }
    __syncthreads();

    int wid = tid >> 5, lane = tid & 31;
    int half = lane >> 4, l15 = lane & 15;
    int o0 = wid * 16;
    f32x8 acc[3] = {};
    for (int kk = 0; kk < CK; kk += 32) {
        __builtin_prefetch(Wb + (((size_t)(o0 + l15)) << ckshift) + kk + 64, 0, 1);
        bf16x16 bf = load_bfrag(Wb, 1 << 30, CK, o0 + l15, kk, lane);
#pragma unroll
        for (int jt = 0; jt < 3; ++jt) {
            if (jt < Jt) {
                bf16x16 af = load_afrag_lds(fA, CK, jt * 16 + l15, kk, half);
                acc[jt] = __builtin_amdgcn_wmma_f32_16x16x32_bf16(
                    false, af, false, bf, (short)0, acc[jt], false, false);
            }
        }
    }
    float best = NEG_INF;
#pragma unroll
    for (int jt = 0; jt < 3; ++jt) {
        if (jt < Jt) {
#pragma unroll
            for (int r = 0; r < 8; ++r) {
                int j = jt * 16 + r + half * 8;
                if (j < K) {
                    float h = acc[jt][r] * BN_INV;
                    h = (h >= 0.f) ? h : SLOPE * h;
                    best = fmaxf(best, h);
                }
            }
        }
    }
    best = fmaxf(best, __shfl_xor(best, 16));
    if (lane < 16)
        catOut[(size_t)rn * catLD + oOff + o0 + l15] = best;
}

// ---------------------------------------------------------------------------
// Generic bf16 WMMA GEMM: Out[M,N] = act(A[M,K] @ Bw[N,K]^T * scale + bias)
// A is f32 (converted on the fly), Bw is preconverted bf16. K % 32 == 0.
// One 16x16 tile per wave; block = 4 waves.
// ---------------------------------------------------------------------------
__global__ void gemm_kernel(const float* __restrict__ A, int lda,
                            const __bf16* __restrict__ Bw,
                            float* __restrict__ Out, int ldo,
                            int M, int N, int K, float scale, int doLrelu,
                            const float* __restrict__ bias) {
    int mt = (M + 15) >> 4, nt = (N + 15) >> 4;
    int wid = threadIdx.x >> 5, lane = threadIdx.x & 31;
    int t = blockIdx.x * (blockDim.x >> 5) + wid;
    if (t >= mt * nt) return;
    int mtile = t / nt, ntile = t - mtile * nt;
    int half = lane >> 4, l15 = lane & 15;
    int m0 = mtile * 16, n0 = ntile * 16;
    f32x8 acc = {};
    for (int kk = 0; kk < K; kk += 32) {
        __builtin_prefetch(Bw + (size_t)(n0 + l15) * K + kk + 64, 0, 1);
        bf16x16 bf = load_bfrag(Bw, N, K, n0 + l15, kk, lane);
        bf16x16 af = load_afrag_f32(A, lda, M, m0 + l15, kk, half);
        acc = __builtin_amdgcn_wmma_f32_16x16x32_bf16(false, af, false, bf,
                                                      (short)0, acc, false, false);
    }
    int ncol = n0 + l15;
    if (ncol < N) {
        float bv = bias ? bias[ncol] : 0.f;
#pragma unroll
        for (int r = 0; r < 8; ++r) {
            int m = m0 + r + half * 8;
            if (m < M) {
                float h = acc[r] * scale + bv;
                if (doLrelu) h = (h >= 0.f) ? h : SLOPE * h;
                Out[(size_t)m * ldo + ncol] = h;
            }
        }
    }
}

// ---------------------------------------------------------------------------
// max/mean pooling over points: g[b][ch] = max_n h, g[b][1024+ch] = mean_n h
// ---------------------------------------------------------------------------
__global__ void reduce_kernel(const float* __restrict__ H, float* __restrict__ g) {
    __shared__ float smax[256], ssum[256];
    int b = blockIdx.y, ch = blockIdx.x, t = threadIdx.x;
    const float* Hb = H + (size_t)b * NPTS * 1024 + ch;
    float mx = NEG_INF, sm = 0.f;
    for (int n = t; n < NPTS; n += 256) {
        float v = Hb[(size_t)n * 1024];
        mx = fmaxf(mx, v);
        sm += v;
    }
    smax[t] = mx; ssum[t] = sm;
    __syncthreads();
    for (int s = 128; s > 0; s >>= 1) {
        if (t < s) { smax[t] = fmaxf(smax[t], smax[t + s]); ssum[t] += ssum[t + s]; }
        __syncthreads();
    }
    if (t == 0) {
        g[(size_t)b * 2048 + ch]        = smax[0];
        g[(size_t)b * 2048 + 1024 + ch] = ssum[0] * (1.0f / NPTS);
    }
}

// ---------------------------------------------------------------------------
// weight conversion f32 [O,Cs] -> bf16 [O,Cp] (zero padded along K)
// ---------------------------------------------------------------------------
__global__ void convw_kernel(const float* __restrict__ W, __bf16* __restrict__ out,
                             int O, int Cs, int Cp) {
    int t = blockIdx.x * blockDim.x + threadIdx.x;
    if (t >= O * Cp) return;
    int o = t / Cp, c = t - o * Cp;
    out[t] = (c < Cs) ? (__bf16)W[(size_t)o * Cs + c] : (__bf16)0.f;
}

// ---------------------------------------------------------------------------
extern "C" void kernel_launch(void* const* d_in, const int* in_sizes, int n_in,
                              void* d_out, int out_size, void* d_ws, size_t ws_size,
                              hipStream_t stream) {
    (void)in_sizes; (void)n_in; (void)out_size; (void)ws_size;
    const float* x    = (const float*)d_in[0];
    const float* bfc3 = (const float*)d_in[17];

    char* ws = (char*)d_ws;
    size_t cur = 0;
    auto alloc = [&](size_t bytes) -> void* {
        void* p = ws + cur;
        cur += (bytes + 255) & ~(size_t)255;
        return p;
    };

    float* D    = (float*)alloc(sizeof(float) * NB * NPTS * NPTS);  // 32MB, reused as H
    float* H    = D;
    float* sqn  = (float*)alloc(sizeof(float) * NROWS);
    int*   knn  = (int*)alloc(sizeof(int) * NROWS * 40);
    float* catb = (float*)alloc(sizeof(float) * (size_t)NROWS * 512);
    float* xcat = (float*)alloc(sizeof(float) * (size_t)NROWS * 512);
    float* g    = (float*)alloc(sizeof(float) * NB * 2048);
    float* h1   = (float*)alloc(sizeof(float) * NB * 512);
    float* h2   = (float*)alloc(sizeof(float) * NB * 256);

    struct WDesc { int in_idx, O, Cs, Cp; __bf16* buf; };
    WDesc wd[13] = {
        {1,   64,   18,   32, nullptr},  // w1s
        {2,   64,   18,   32, nullptr},  // w1l
        {3,   64,  128,  128, nullptr},  // w1f
        {4,   64,  128,  128, nullptr},  // w2s
        {5,   64,  128,  128, nullptr},  // w2l
        {6,   64,  128,  128, nullptr},  // w2f
        {7,  128,  128,  128, nullptr},  // w3s
        {8,  128,  128,  128, nullptr},  // w3l
        {9,  128,  256,  256, nullptr},  // w3f
        {10, 256,  256,  256, nullptr},  // w4s
        {11, 256,  256,  256, nullptr},  // w4l
        {12, 256,  512,  512, nullptr},  // w4f
        {13, 1024, 512,  512, nullptr},  // wc
    };
    for (int i = 0; i < 13; ++i)
        wd[i].buf = (__bf16*)alloc(sizeof(__bf16) * (size_t)wd[i].O * wd[i].Cp);
    __bf16* wfc1b = (__bf16*)alloc(sizeof(__bf16) * 512 * 2048);
    __bf16* wfc2b = (__bf16*)alloc(sizeof(__bf16) * 256 * 512);
    __bf16* wfc3b = (__bf16*)alloc(sizeof(__bf16) * 40 * 256);

    for (int i = 0; i < 13; ++i) {
        int tot = wd[i].O * wd[i].Cp;
        convw_kernel<<<(tot + 255) / 256, 256, 0, stream>>>(
            (const float*)d_in[wd[i].in_idx], wd[i].buf, wd[i].O, wd[i].Cs, wd[i].Cp);
    }
    convw_kernel<<<(512 * 2048 + 255) / 256, 256, 0, stream>>>((const float*)d_in[14], wfc1b, 512, 2048, 2048);
    convw_kernel<<<(256 * 512 + 255) / 256, 256, 0, stream>>>((const float*)d_in[15], wfc2b, 256, 512, 512);
    convw_kernel<<<(40 * 256 + 255) / 256, 256, 0, stream>>>((const float*)d_in[16], wfc3b, 40, 256, 256);

    auto run_layer = [&](const float* Xin, int ldx, int C,
                         const __bf16* Ws, const __bf16* Wl, int O, int ckshift,
                         const __bf16* Wf, int Of, int chOff) {
        int CK = 1 << ckshift;
        sqn_kernel<<<(NROWS + 255) / 256, 256, 0, stream>>>(Xin, ldx, C, sqn);
        dim3 dg(16, 64, NB);
        dist_kernel<<<dg, 128, 0, stream>>>(Xin, ldx, C, sqn, D);
        topk_kernel<<<NROWS / 8, 256, 0, stream>>>(D, knn);
        size_t hdrBytes = (((size_t)C * 4 + 192 + 63) & ~(size_t)63);
        {   // small scale k=20 -> 2 j-tiles
            size_t sm = hdrBytes + (size_t)2 * 16 * CK * sizeof(__bf16);
            edgeconv_kernel<<<NROWS, (O / 16) * 32, sm, stream>>>(
                Xin, ldx, C, knn, 20, 2, Ws, ckshift, catb, 2 * O, 0);
        }
        {   // large scale k=40 -> 3 j-tiles
            size_t sm = hdrBytes + (size_t)3 * 16 * CK * sizeof(__bf16);
            edgeconv_kernel<<<NROWS, (O / 16) * 32, sm, stream>>>(
                Xin, ldx, C, knn, 40, 3, Wl, ckshift, catb, 2 * O, O);
        }
        int tiles = ((NROWS + 15) / 16) * ((Of + 15) / 16);
        gemm_kernel<<<(tiles + 3) / 4, 128, 0, stream>>>(
            catb, 2 * O, Wf, xcat + chOff, 512, NROWS, Of, 2 * O, BN_INV, 1, nullptr);
    };

    run_layer(x,          9,   9,   wd[0].buf, wd[1].buf,  64, 5, wd[2].buf,  64,   0);
    run_layer(xcat + 0,   512, 64,  wd[3].buf, wd[4].buf,  64, 7, wd[5].buf,  64,  64);
    run_layer(xcat + 64,  512, 64,  wd[6].buf, wd[7].buf, 128, 7, wd[8].buf, 128, 128);
    run_layer(xcat + 128, 512, 128, wd[9].buf, wd[10].buf, 256, 8, wd[11].buf, 256, 256);

    // h = lrelu(xcat @ wc^T * BN_INV)  -> H (reuses D space)
    {
        int tiles = (NROWS / 16) * (1024 / 16);
        gemm_kernel<<<(tiles + 3) / 4, 128, 0, stream>>>(
            xcat, 512, wd[12].buf, H, 1024, NROWS, 1024, 512, BN_INV, 1, nullptr);
    }
    // g = [max_n h, mean_n h]
    {
        dim3 rg(1024, NB);
        reduce_kernel<<<rg, 256, 0, stream>>>(H, g);
    }
    // FC chain (M=8, padded inside one M-tile)
    {
        int t1 = 1 * (512 / 16);
        gemm_kernel<<<(t1 + 3) / 4, 128, 0, stream>>>(g, 2048, wfc1b, h1, 512, NB, 512, 2048, BN_INV, 1, nullptr);
        int t2 = 1 * (256 / 16);
        gemm_kernel<<<(t2 + 3) / 4, 128, 0, stream>>>(h1, 512, wfc2b, h2, 256, NB, 256, 512, BN_INV, 1, nullptr);
        int t3 = 1 * ((40 + 15) / 16);
        gemm_kernel<<<(t3 + 3) / 4, 128, 0, stream>>>(h2, 256, wfc3b, (float*)d_out, 40, NB, 40, 256, 1.0f, 0, bfc3);
    }
}
